// SGLCEncoder_27118423507258
// MI455X (gfx1250) — matrix operand order repaired and checked by reading, not compiled, not run on hardware
//
#include <hip/hip_runtime.h>

#define T_STEPS 64
#define BATCH   32
#define NNODE   64
#define DIN_    128
#define HID     128

typedef __attribute__((ext_vector_type(16))) __bf16 v16bf;
typedef __attribute__((ext_vector_type(8)))  __bf16 v8bf;
typedef __attribute__((ext_vector_type(8)))  float  v8f;

union V16u { v16bf v; v8bf h[2]; };

// A operand (16x32 bf16, row-major source, contiguous lane layout):
// lanes 0-15: row m0+l, K = k0+0..15 ; lanes 16-31: K = k0+16..31
__device__ __forceinline__ v16bf load_a_bf(const __bf16* base, int ldk, int m0, int k0, int l, int hi) {
  const __bf16* p = base + (m0 + l) * ldk + k0 + hi * 16;
  V16u u;
  u.h[0] = *(const v8bf*)p;
  u.h[1] = *(const v8bf*)(p + 8);
  return u.v;
}

// B operand from B^T stored row-major (N x K): lanes 0-15 get K=k0..k0+15 of col n0+l,
// lanes 16-31 get K=k0+16..k0+31 -> one contiguous 32B load per lane.
__device__ __forceinline__ v16bf load_bt_bf(const __bf16* bt, int ldk, int n0, int k0, int l, int hi) {
  return *(const v16bf*)(bt + (n0 + l) * ldk + k0 + hi * 16);
}

// segmented A over k: [0,128): a0 ld128, [128,256): a1 ld128, [256,512): a2 ld256
__device__ __forceinline__ v16bf load_a_seg3(const __bf16* a0, const __bf16* a1, const __bf16* a2,
                                             int m0, int k0, int l, int hi) {
  if (k0 < 128) return load_a_bf(a0, 128, m0, k0, l, hi);
  if (k0 < 256) return load_a_bf(a1, 128, m0, k0 - 128, l, hi);
  return load_a_bf(a2, 256, m0, k0 - 256, l, hi);
}

// conv B operand: feature d < 128 -> x^T, else h^T (or (r*h)^T)
__device__ __forceinline__ v16bf load_conv_b(const __bf16* XT, const __bf16* HT,
                                             int n, int k0, int l, int hi) {
  return (n < 128) ? load_bt_bf(XT, 64, n, k0, l, hi)
                   : load_bt_bf(HT, 64, n - 128, k0, l, hi);
}

__device__ __forceinline__ v8f wmma_bf16(v16bf a, v16bf b, v8f c) {
  return __builtin_amdgcn_wmma_f32_16x16x32_bf16(false, a, false, b, (short)0, c, false, false);
}

// Async prefetch of one (N x 128) f32 x-slab into LDS: each of 512 threads copies
// 4 x 16B via GLOBAL_LOAD_ASYNC_TO_LDS_B128 (ASYNCcnt-tracked, no VGPR data path).
__device__ __forceinline__ void prefetch_x_async(const float* xp, unsigned lds_off, int tid) {
#pragma unroll
  for (int j = 0; j < 4; ++j) {
    unsigned la = lds_off + (unsigned)(tid * 16 + j * 8192);
    unsigned long long ga = (unsigned long long)(xp + tid * 4 + j * 2048);
    asm volatile("global_load_async_to_lds_b128 %0, %1, off" :: "v"(la), "v"(ga) : "memory");
  }
}

__global__ void transpose_cast_kernel(const float* __restrict__ src, __bf16* __restrict__ dst,
                                      int K, int N) {
  int i = blockIdx.x * 256 + threadIdx.x;
  if (i < K * N) {
    int k = i / N, n = i - k * N;
    dst[n * K + k] = (__bf16)src[i];
  }
}

// One workgroup per batch element; internal scan over T.
__global__ void __launch_bounds__(512)
sglc_layer_kernel(const float* __restrict__ xsrc,     // (T,B,N*128)
                  const float* __restrict__ h0,       // (B,N*H)
                  const float* __restrict__ sup_in,   // (B,N,N) f32
                  const __bf16* __restrict__ Wgt, const float* __restrict__ bg,   // Wg^T (256x512)
                  const __bf16* __restrict__ Wct, const float* __restrict__ bc,   // Wc^T (128x512)
                  const __bf16* __restrict__ Wqt, const __bf16* __restrict__ Wkt, // (128x128)
                  float* __restrict__ hs_out,         // (T,B,N*H)
                  float* __restrict__ hlast_out,      // (B,N*H)
                  float* __restrict__ sup_out)        // (B,N,N) f32 handoff
{
  extern __shared__ char smem[];
  __bf16* SUP  = (__bf16*)(smem + 0);        // 64x64   (8 KB)   persistent across steps
  __bf16* H    = (__bf16*)(smem + 8192);     // 64x128  (16 KB)  persistent hidden state
  __bf16* X    = (__bf16*)(smem + 24576);    // 64x128  (16 KB)  row-major x ; later Q
  __bf16* XT   = (__bf16*)(smem + 40960);    // 128x64  (16 KB)  x^T ; later Kmat
  __bf16* HT   = (__bf16*)(smem + 57344);    // 128x64  (16 KB)  h^T ; later (r*h)^T ; later SC
  __bf16* CONV = (__bf16*)(smem + 73728);    // 64x256  (32 KB)  conv, then conv_c
  __bf16* RH   = (__bf16*)(smem + 106496);   // 64x128  (16 KB)  r*h row-major
  __bf16* U    = (__bf16*)(smem + 122880);   // 64x128  (16 KB)  update gate
  float*  XN   = (float*)(smem + 139264);    // 64x128 f32 (32 KB) async x staging
  float*  SC   = (float*)(smem + 57344);     // 64x64 f32 (overlays HT after it is dead)
  __bf16* Q    = X;
  __bf16* Km   = XT;

  const int tid  = threadIdx.x;
  const int b    = blockIdx.x;
  const int wv   = tid >> 5;
  const int lane = tid & 31;
  const int l    = lane & 15;
  const int hi   = lane >> 4;
  const unsigned xn_off = (unsigned)(size_t)(void*)XN;   // LDS byte address of XN

  // prologue: kick off async prefetch of x_0, overlap with state init
  prefetch_x_async(xsrc + (size_t)b * (NNODE * DIN_), xn_off, tid);

  for (int i = tid; i < NNODE * HID; i += 512)   H[i]   = (__bf16)h0[b * NNODE * HID + i];
  for (int i = tid; i < NNODE * NNODE; i += 512) SUP[i] = (__bf16)sup_in[b * NNODE * NNODE + i];
  __syncthreads();

  for (int t = 0; t < T_STEPS; ++t) {
    // ---- phase 1: wait for async x, stage x, x^T, h^T; then prefetch x_{t+1} ----
    asm volatile("s_wait_asynccnt 0x0" ::: "memory");
    __syncthreads();
    for (int i = tid; i < NNODE * DIN_; i += 512) {
      int m = i >> 7, d = i & 127;
      float v = XN[i];
      X[i] = (__bf16)v;
      XT[d * 64 + m] = (__bf16)v;
      HT[d * 64 + m] = H[m * 128 + d];
    }
    __syncthreads();
    if (t + 1 < T_STEPS)
      prefetch_x_async(xsrc + ((size_t)(t + 1) * BATCH + b) * (NNODE * DIN_), xn_off, tid);

    // ---- phase 2: conv = sup @ [x|h]  (64x256, K=64); 4 tiles/wave, shared A ----
    {
      int m0 = (wv >> 2) * 16;
      int n0 = (wv & 3) * 64;
      v8f z = {};
      v8f acc[4] = {z, z, z, z};
      for (int kk = 0; kk < 64; kk += 32) {
        v16bf av = load_a_bf(SUP, 64, m0, kk, l, hi);
#pragma unroll
        for (int i2 = 0; i2 < 4; ++i2)
          acc[i2] = wmma_bf16(av, load_conv_b(XT, HT, n0 + 16 * i2, kk, l, hi), acc[i2]);
      }
#pragma unroll
      for (int i2 = 0; i2 < 4; ++i2)
        for (int r = 0; r < 8; ++r)
          CONV[(m0 + r + hi * 8) * 256 + n0 + 16 * i2 + l] = (__bf16)acc[i2][r];
    }
    __syncthreads();

    // ---- phase 3: gates = sigmoid([x|h|conv] @ Wg + bg)  (64x256, K=512) ----
    {
      int m0 = (wv >> 2) * 16;
      int n0 = (wv & 3) * 64;
      v8f z = {};
      v8f acc[4] = {z, z, z, z};
      for (int kk = 0; kk < 512; kk += 32) {
        v16bf av = load_a_seg3(X, H, CONV, m0, kk, l, hi);
#pragma unroll
        for (int i2 = 0; i2 < 4; ++i2)
          acc[i2] = wmma_bf16(av, load_bt_bf(Wgt, 512, n0 + 16 * i2, kk, l, hi), acc[i2]);
      }
#pragma unroll
      for (int i2 = 0; i2 < 4; ++i2) {
        for (int r = 0; r < 8; ++r) {
          int m = m0 + r + hi * 8, n = n0 + 16 * i2 + l;
          float g = 1.f / (1.f + __expf(-(acc[i2][r] + bg[n])));
          if (n < 128) {                       // reset gate -> build r*h and (r*h)^T
            float rh = g * (float)H[m * 128 + n];
            RH[m * 128 + n] = (__bf16)rh;
            HT[n * 64 + m]  = (__bf16)rh;      // HT now holds (r*h)^T
          } else {                             // update gate
            U[m * 128 + (n - 128)] = (__bf16)g;
          }
        }
      }
    }
    __syncthreads();

    // ---- phase 4: conv_c = sup @ [x | r*h]  (64x256, K=64) ----
    {
      int m0 = (wv >> 2) * 16;
      int n0 = (wv & 3) * 64;
      v8f z = {};
      v8f acc[4] = {z, z, z, z};
      for (int kk = 0; kk < 64; kk += 32) {
        v16bf av = load_a_bf(SUP, 64, m0, kk, l, hi);
#pragma unroll
        for (int i2 = 0; i2 < 4; ++i2)
          acc[i2] = wmma_bf16(av, load_conv_b(XT, HT, n0 + 16 * i2, kk, l, hi), acc[i2]);
      }
#pragma unroll
      for (int i2 = 0; i2 < 4; ++i2)
        for (int r = 0; r < 8; ++r)
          CONV[(m0 + r + hi * 8) * 256 + n0 + 16 * i2 + l] = (__bf16)acc[i2][r];
    }
    __syncthreads();

    // ---- phase 5: c = tanh([x|r*h|conv_c] @ Wc + bc); h_new = u*h + (1-u)*c ----
    {
      int m0 = (wv >> 2) * 16;
      int n0 = (wv & 3) * 32;
      v8f z = {};
      v8f acc[2] = {z, z};
      for (int kk = 0; kk < 512; kk += 32) {
        v16bf av = load_a_seg3(X, RH, CONV, m0, kk, l, hi);
#pragma unroll
        for (int i2 = 0; i2 < 2; ++i2)
          acc[i2] = wmma_bf16(av, load_bt_bf(Wct, 512, n0 + 16 * i2, kk, l, hi), acc[i2]);
      }
#pragma unroll
      for (int i2 = 0; i2 < 2; ++i2) {
        for (int r = 0; r < 8; ++r) {
          int m = m0 + r + hi * 8, n = n0 + 16 * i2 + l;
          float c  = tanhf(acc[i2][r] + bc[n]);
          float u_ = (float)U[m * 128 + n];
          float h_ = (float)H[m * 128 + n];
          float hn = u_ * h_ + (1.f - u_) * c;
          H[m * 128 + n] = (__bf16)hn;                       // in-place state update
          hs_out[((size_t)t * BATCH + b) * 8192 + m * 128 + n] = hn;
          if (t == T_STEPS - 1)
            hlast_out[b * 8192 + m * 128 + n] = hn;
        }
      }
    }
    __syncthreads();

    // ---- phase 6: q = h_new @ Wq, k = h_new @ Wk  (each 64x128, K=128) ----
    {
      int isK = wv >> 3;
      int w8  = wv & 7;
      int m0  = (w8 >> 1) * 16;
      int n0  = (w8 & 1) * 64;
      const __bf16* Wt = isK ? Wkt : Wqt;
      v8f z = {};
      v8f acc[4] = {z, z, z, z};
      for (int kk = 0; kk < 128; kk += 32) {
        v16bf av = load_a_bf(H, 128, m0, kk, l, hi);
#pragma unroll
        for (int i2 = 0; i2 < 4; ++i2)
          acc[i2] = wmma_bf16(av, load_bt_bf(Wt, 128, n0 + 16 * i2, kk, l, hi), acc[i2]);
      }
      __bf16* Dst = isK ? Km : Q;
#pragma unroll
      for (int i2 = 0; i2 < 4; ++i2)
        for (int r = 0; r < 8; ++r)
          Dst[(m0 + r + hi * 8) * 128 + n0 + 16 * i2 + l] = (__bf16)acc[i2][r];
    }
    __syncthreads();

    // ---- phase 7: scores = q @ k^T  (64x64, K=128); B^T of k^T is k row-major ----
    {
      int m0 = (wv >> 2) * 16, n0 = (wv & 3) * 16;
      v8f acc = {};
      for (int kk = 0; kk < 128; kk += 32) {
        v16bf av = load_a_bf(Q, 128, m0, kk, l, hi);
        v16bf bm = load_bt_bf(Km, 128, n0, kk, l, hi);
        acc = wmma_bf16(av, bm, acc);
      }
      for (int r = 0; r < 8; ++r)
        SC[(m0 + r + hi * 8) * 64 + n0 + l] = acc[r];
    }
    __syncthreads();

    // ---- phase 8: sup = softmax(relu(scores), axis=-1) ----
    if (tid < 64) {
      float mx = 0.f;
      for (int j2 = 0; j2 < 64; ++j2) {
        float v = SC[tid * 64 + j2];
        v = v > 0.f ? v : 0.f;
        mx = v > mx ? v : mx;
      }
      float s = 0.f;
      for (int j2 = 0; j2 < 64; ++j2) {
        float v = SC[tid * 64 + j2];
        v = v > 0.f ? v : 0.f;
        s += __expf(v - mx);
      }
      float inv = 1.f / s;
      for (int j2 = 0; j2 < 64; ++j2) {
        float v = SC[tid * 64 + j2];
        v = v > 0.f ? v : 0.f;
        float p = __expf(v - mx) * inv;
        SUP[tid * 64 + j2] = (__bf16)p;
        if (t == T_STEPS - 1)
          sup_out[b * 4096 + tid * 64 + j2] = p;   // layer handoff (f32)
      }
    }
    __syncthreads();
  }
}

extern "C" void kernel_launch(void* const* d_in, const int* in_sizes, int n_in,
                              void* d_out, int out_size, void* d_ws, size_t ws_size,
                              hipStream_t stream) {
  const float* inputs = (const float*)d_in[0];
  const float* h0     = (const float*)d_in[1];
  const float* sup    = (const float*)d_in[2];
  const float* Wg0 = (const float*)d_in[3];  const float* bg0 = (const float*)d_in[4];
  const float* Wc0 = (const float*)d_in[5];  const float* bc0 = (const float*)d_in[6];
  const float* Wq0 = (const float*)d_in[7];  const float* Wk0 = (const float*)d_in[8];
  const float* Wg1 = (const float*)d_in[9];  const float* bg1 = (const float*)d_in[10];
  const float* Wc1 = (const float*)d_in[11]; const float* bc1 = (const float*)d_in[12];
  const float* Wq1 = (const float*)d_in[13]; const float* Wk1 = (const float*)d_in[14];

  char* ws = (char*)d_ws;
  __bf16* Wgt0 = (__bf16*)(ws + 0);        // 256x512
  __bf16* Wct0 = (__bf16*)(ws + 262144);   // 128x512
  __bf16* Wqt0 = (__bf16*)(ws + 393216);   // 128x128
  __bf16* Wkt0 = (__bf16*)(ws + 425984);
  __bf16* Wgt1 = (__bf16*)(ws + 458752);
  __bf16* Wct1 = (__bf16*)(ws + 720896);
  __bf16* Wqt1 = (__bf16*)(ws + 851968);
  __bf16* Wkt1 = (__bf16*)(ws + 884736);
  float*  sup_h = (float*)(ws + 917504);   // (B,N,N) f32 layer handoff

  float* out_hidden = (float*)d_out;                       // (2, B, 8192)
  float* cur = out_hidden + 2 * BATCH * 8192;              // (T, B, 8192)

  // weight transpose+cast to bf16 (B-operand friendly N x K layout)
  {
    int n;
    n = 512 * 256; transpose_cast_kernel<<<(n + 255) / 256, 256, 0, stream>>>(Wg0, Wgt0, 512, 256);
    n = 512 * 128; transpose_cast_kernel<<<(n + 255) / 256, 256, 0, stream>>>(Wc0, Wct0, 512, 128);
    n = 128 * 128; transpose_cast_kernel<<<(n + 255) / 256, 256, 0, stream>>>(Wq0, Wqt0, 128, 128);
    n = 128 * 128; transpose_cast_kernel<<<(n + 255) / 256, 256, 0, stream>>>(Wk0, Wkt0, 128, 128);
    n = 512 * 256; transpose_cast_kernel<<<(n + 255) / 256, 256, 0, stream>>>(Wg1, Wgt1, 512, 256);
    n = 512 * 128; transpose_cast_kernel<<<(n + 255) / 256, 256, 0, stream>>>(Wc1, Wct1, 512, 128);
    n = 128 * 128; transpose_cast_kernel<<<(n + 255) / 256, 256, 0, stream>>>(Wq1, Wqt1, 128, 128);
    n = 128 * 128; transpose_cast_kernel<<<(n + 255) / 256, 256, 0, stream>>>(Wk1, Wkt1, 128, 128);
  }

  const size_t smem = 172032;  // 168 KB dynamic LDS (gfx1250: up to 320 KB/WG)

  // layer 0: x from inputs, sup from d_in; hs -> cur region (staging for layer 1)
  sglc_layer_kernel<<<BATCH, 512, smem, stream>>>(
      inputs, h0, sup, Wgt0, bg0, Wct0, bc0, Wqt0, Wkt0,
      cur, out_hidden, sup_h);

  // layer 1: x from cur (read-then-overwrite per timestep), sup from layer-0 handoff
  sglc_layer_kernel<<<BATCH, 512, smem, stream>>>(
      cur, h0 + BATCH * 8192, sup_h, Wgt1, bg1, Wct1, bc1, Wqt1, Wkt1,
      cur, out_hidden + BATCH * 8192, sup_h);
}
